// KalmanCTRNN_46866683134475
// MI455X (gfx1250) — compile-verified, matrix operand order checked by reading, not run on hardware
//
#include <hip/hip_runtime.h>

#define T_STEPS 1000
#define BATCH   512
#define IN_DIM  32
#define HD      64
#define OBS     3
#define NHIST   5
#define ALPHA   0.2f

typedef __attribute__((ext_vector_type(16))) __bf16 v16bf;
typedef __attribute__((ext_vector_type(2)))  __bf16 v2bf;
typedef __attribute__((ext_vector_type(8)))  float  v8f;
typedef __attribute__((ext_vector_type(2)))  float  v2f;

#if __has_builtin(__builtin_amdgcn_wmma_f32_16x16x4_f32)
#define KC_USE_WMMA 1
#else
#define KC_USE_WMMA 0
#endif

// ---------------------------------------------------------------------------
// Kernel A: sequential Riccati recursion (data-independent of x / hidden).
// Produces KC_t (bf16, [T][64][64]) into workspace; KC_t = 0 for t < NHIST.
// Single workgroup (8 waves). The three 64x64x64 matmuls per step run on
// V_WMMA_F32_16X16X4_F32 (fp32 matrix pipe -> full precision for the 1000-step
// recursion); W_hh / W_hh^T fragments are register-resident across all steps.
// Falls back to scalar FMA if the f32 WMMA builtin is unavailable.
// ---------------------------------------------------------------------------
__global__ void __launch_bounds__(256)
kc_precompute(const float* __restrict__ Whh_g, const float* __restrict__ Cm_g,
              unsigned short* __restrict__ KCout_raw) {
  __bf16* __restrict__ KCout = (__bf16*)KCout_raw;
  __shared__ __align__(16) float A[HD * HD];      // W_hh
  __shared__ __align__(16) float Cm[OBS * HD];
  __shared__ __align__(16) float cov[HD * HD];
  __shared__ __align__(16) float P[HD * HD];
  __shared__ __align__(16) float M1[HD * HD];
  __shared__ __align__(16) float KC[HD * HD];
  __shared__ __align__(16) float CP[OBS * HD];
  __shared__ __align__(16) float PCt[HD * OBS];
  __shared__ __align__(16) float Sm[OBS * OBS];
  __shared__ __align__(16) float Sinv[OBS * OBS];

  const int tid  = threadIdx.x;
  const int wave = tid >> 5;
  const int lane = tid & 31;
  const int half = lane >> 4;
  const int ln   = lane & 15;
  const int tr   = wave >> 1;         // output row tile (0..3)
  const int tc0  = (wave & 1) * 2;    // first of two output col tiles

  for (int i = tid; i < HD * HD; i += 256) {
    A[i] = Whh_g[i];
    cov[i] = ((i >> 6) == (i & 63)) ? 1.f : 0.f;
  }
  for (int i = tid; i < OBS * HD; i += 256) Cm[i] = Cm_g[i];
  __syncthreads();

#if KC_USE_WMMA
  // Constant fragments of W_hh: A-frags (row tile tr) and B-frags of W_hh^T
  // (col tiles tc0, tc0+1).  f32 16x16x4 layout: element i <-> k = 2*half + i.
  v2f afA[16];
  v2f bfAT[2][16];
#pragma unroll
  for (int ks = 0; ks < 16; ++ks) {
    int kg = 4 * ks + 2 * half;
    afA[ks]     = *(const v2f*)&A[(16 * tr + ln) * HD + kg];
    bfAT[0][ks] = *(const v2f*)&A[(16 * tc0 + ln) * HD + kg];
    bfAT[1][ks] = *(const v2f*)&A[(16 * (tc0 + 1) + ln) * HD + kg];
  }
#endif

  for (int t = 0; t < T_STEPS; ++t) {
    if (t < NHIST) {  // uniform branch: no correction, KC = 0, cov unchanged
      for (int i = tid; i < HD * HD; i += 256)
        KCout[(size_t)t * HD * HD + i] = (__bf16)0.f;
      continue;
    }
#if KC_USE_WMMA
    // ---- M1 = A @ cov  (fp32 WMMA) ----
#pragma unroll
    for (int q = 0; q < 2; ++q) {
      const int tc = tc0 + q;
      v8f acc = {};
#pragma unroll
      for (int ks = 0; ks < 16; ++ks) {
        int kg = 4 * ks + 2 * half;
        v2f b;
        b[0] = cov[(kg + 0) * HD + 16 * tc + ln];
        b[1] = cov[(kg + 1) * HD + 16 * tc + ln];
        acc = __builtin_amdgcn_wmma_f32_16x16x4_f32(false, afA[ks], false, b,
                                                    (short)0, acc, false, false);
      }
#pragma unroll
      for (int r = 0; r < 8; ++r)
        M1[(16 * tr + r + 8 * half) * HD + 16 * tc + ln] = acc[r];
    }
    __syncthreads();
    // ---- P = M1 @ A^T  (fp32 WMMA, W_hh^T frags in registers) ----
#pragma unroll
    for (int q = 0; q < 2; ++q) {
      const int tc = tc0 + q;
      v8f acc = {};
#pragma unroll
      for (int ks = 0; ks < 16; ++ks) {
        int kg = 4 * ks + 2 * half;
        v2f a = *(const v2f*)&M1[(16 * tr + ln) * HD + kg];
        acc = __builtin_amdgcn_wmma_f32_16x16x4_f32(false, a, false, bfAT[q][ks],
                                                    (short)0, acc, false, false);
      }
#pragma unroll
      for (int r = 0; r < 8; ++r)
        P[(16 * tr + r + 8 * half) * HD + 16 * tc + ln] = acc[r];
    }
    __syncthreads();
#else
    for (int i = tid; i < HD * HD; i += 256) {
      int r = i >> 6, c = i & 63;
      float s = 0.f;
      for (int k = 0; k < HD; ++k) s += A[r * HD + k] * cov[k * HD + c];
      M1[i] = s;
    }
    __syncthreads();
    for (int i = tid; i < HD * HD; i += 256) {
      int r = i >> 6, c = i & 63;
      float s = 0.f;
      for (int k = 0; k < HD; ++k) s += M1[r * HD + k] * A[c * HD + k];
      P[i] = s;
    }
    __syncthreads();
#endif
    // ---- CP = C @ P (3x64), PCt = P @ C^T (64x3) : tiny, scalar ----
    for (int i = tid; i < OBS * HD; i += 256) {
      int o = i / HD, c = i % HD;
      float s = 0.f;
      for (int k = 0; k < HD; ++k) s += Cm[o * HD + k] * P[k * HD + c];
      CP[i] = s;
    }
    for (int i = tid; i < HD * OBS; i += 256) {
      int r = i / OBS, o = i % OBS;
      float s = 0.f;
      for (int k = 0; k < HD; ++k) s += P[r * HD + k] * Cm[o * HD + k];
      PCt[i] = s;
    }
    __syncthreads();
    // ---- S = CP @ C^T + I3 ----
    if (tid < OBS * OBS) {
      int r = tid / OBS, c = tid % OBS;
      float s = (r == c) ? 1.f : 0.f;
      for (int k = 0; k < HD; ++k) s += CP[r * HD + k] * Cm[c * HD + k];
      Sm[tid] = s;
    }
    __syncthreads();
    if (tid == 0) {  // 3x3 inverse via adjugate
      float a = Sm[0], b = Sm[1], c = Sm[2], d = Sm[3], e = Sm[4], f = Sm[5],
            g = Sm[6], h = Sm[7], i2 = Sm[8];
      float A0 = e * i2 - f * h, B0 = -(d * i2 - f * g), C0 = d * h - e * g;
      float inv = 1.f / (a * A0 + b * B0 + c * C0);
      Sinv[0] = A0 * inv;               Sinv[1] = (c * h - b * i2) * inv;
      Sinv[2] = (b * f - c * e) * inv;  Sinv[3] = B0 * inv;
      Sinv[4] = (a * i2 - c * g) * inv; Sinv[5] = (c * d - a * f) * inv;
      Sinv[6] = C0 * inv;               Sinv[7] = (b * g - a * h) * inv;
      Sinv[8] = (a * e - b * d) * inv;
    }
    __syncthreads();
    // ---- KC = (P C^T S^-1) C : rank-3, scalar ----
    for (int i = tid; i < HD * HD; i += 256) {
      int r = i >> 6, c = i & 63;
      float s = 0.f;
      for (int o = 0; o < OBS; ++o) {
        float kro = PCt[r * OBS + 0] * Sinv[0 * OBS + o] +
                    PCt[r * OBS + 1] * Sinv[1 * OBS + o] +
                    PCt[r * OBS + 2] * Sinv[2 * OBS + o];
        s += kro * Cm[o * HD + c];
      }
      KC[i] = s;
      KCout[(size_t)t * HD * HD + i] = (__bf16)s;
    }
    __syncthreads();
    // ---- cov = P - KC @ P ----
#if KC_USE_WMMA
#pragma unroll
    for (int q = 0; q < 2; ++q) {
      const int tc = tc0 + q;
      v8f acc = {};
#pragma unroll
      for (int ks = 0; ks < 16; ++ks) {
        int kg = 4 * ks + 2 * half;
        v2f a = *(const v2f*)&KC[(16 * tr + ln) * HD + kg];
        v2f b;
        b[0] = P[(kg + 0) * HD + 16 * tc + ln];
        b[1] = P[(kg + 1) * HD + 16 * tc + ln];
        acc = __builtin_amdgcn_wmma_f32_16x16x4_f32(false, a, false, b,
                                                    (short)0, acc, false, false);
      }
#pragma unroll
      for (int r = 0; r < 8; ++r) {
        int idx = (16 * tr + r + 8 * half) * HD + 16 * tc + ln;
        cov[idx] = P[idx] - acc[r];
      }
    }
#else
    for (int i = tid; i < HD * HD; i += 256) {
      int r = i >> 6, c = i & 63;
      float s = 0.f;
      for (int k = 0; k < HD; ++k) s += KC[r * HD + k] * P[k * HD + c];
      cov[i] = P[i] - s;
    }
#endif
    __syncthreads();
  }
}

// ---------------------------------------------------------------------------
// Kernel B: persistent per-batch-tile recurrence. Each WG owns 16 batch rows,
// 4 waves each own a 16-wide N-slice of H. bf16 WMMA (fp32 accum) for all
// GEMMs; weight B-fragments live in VGPRs for all 1000 steps; 5-deep LDS
// ring holds h_obs history (fp32 master + bf16 mirror for fragment builds).
// Ring-slot indices are maintained incrementally (no div/mod in hot loop);
// next-step x / KC lines are prefetched (global_prefetch_b8).
// ---------------------------------------------------------------------------
__global__ void __launch_bounds__(128)
ctrnn_main(const float* __restrict__ x,
           const float* __restrict__ W_ih, const float* __restrict__ b_ih,
           const float* __restrict__ W_hh, const float* __restrict__ b_hh,
           const float* __restrict__ W_mlp, const float* __restrict__ b_mlp,
           const unsigned short* __restrict__ KC_raw,
           float* __restrict__ out, float* __restrict__ hlast) {
  const __bf16* __restrict__ KCbf = (const __bf16*)KC_raw;

  __shared__ __align__(16) float  ring32[NHIST][16][HD];  // fp32 h_obs ring
  __shared__ __align__(16) __bf16 ringbf[NHIST][16][HD];  // bf16 mirror
  __shared__ __align__(16) __bf16 xbf[16][IN_DIM];        // x tile, bf16
  __shared__ __align__(16) __bf16 dtile[16][HD];          // (h_mlp - h_obs)

  const int tid  = threadIdx.x;
  const int wave = tid >> 5;       // 0..3 -> output column tile
  const int lane = tid & 31;
  const int half = lane >> 4;      // K-half selector in 16-bit frag layout
  const int ln   = lane & 15;      // row m (A) or col n (B / C-D)
  const int nb   = wave * 16;
  const int rowbase = blockIdx.x * 16;

  for (int i = tid; i < NHIST * 16 * HD; i += 128) {
    (&ring32[0][0][0])[i] = 0.f;
    (&ringbf[0][0][0])[i] = (__bf16)0.f;
  }

  // 16-bit A/B fragment K pattern: element pair p -> k = (p>=4)*16 + half*8 + 2*(p&3)
  int kpair[8];
#pragma unroll
  for (int p = 0; p < 8; ++p) kpair[p] = ((p >= 4) ? 16 : 0) + (half << 3) + ((p & 3) << 1);

  // ---- one-time: weight B fragments into registers ----
  v16bf Bih;
  v16bf Bhh[2];
  v16bf Bmlp[10];
  {
    const int ncol = nb + ln;
#pragma unroll
    for (int p = 0; p < 8; ++p) {
      int k = kpair[p];
      Bih[2 * p]     = (__bf16)W_ih[ncol * IN_DIM + k];
      Bih[2 * p + 1] = (__bf16)W_ih[ncol * IN_DIM + k + 1];
    }
#pragma unroll
    for (int kt = 0; kt < 2; ++kt)
#pragma unroll
      for (int p = 0; p < 8; ++p) {
        int k = kt * 32 + kpair[p];
        Bhh[kt][2 * p]     = (__bf16)W_hh[ncol * HD + k];
        Bhh[kt][2 * p + 1] = (__bf16)W_hh[ncol * HD + k + 1];
      }
#pragma unroll
    for (int f = 0; f < 10; ++f)
#pragma unroll
      for (int p = 0; p < 8; ++p) {
        int k = f * 32 + kpair[p];
        Bmlp[f][2 * p]     = (__bf16)W_mlp[ncol * (NHIST * HD) + k];
        Bmlp[f][2 * p + 1] = (__bf16)W_mlp[ncol * (NHIST * HD) + k + 1];
      }
  }
  const float biasH = b_ih[nb + ln] + b_hh[nb + ln];
  const float biasM = b_mlp[nb + ln];
  __syncthreads();

  int wslot = 0;  // == t % NHIST, maintained incrementally
  for (int t = 0; t < T_STEPS; ++t) {
    const int hslot = (wslot == 0) ? (NHIST - 1) : (wslot - 1);  // (t-1)%5

    // ---- phase 0: cooperative x tile load (coalesced float4) -> bf16 LDS,
    //              plus prefetch of next step's x tile and KC_t+1
    {
      int r = tid >> 3;
      int c = (tid & 7) << 2;
      const float4 v =
          *(const float4*)&x[(size_t)t * BATCH * IN_DIM + (size_t)(rowbase + r) * IN_DIM + c];
      xbf[r][c + 0] = (__bf16)v.x;
      xbf[r][c + 1] = (__bf16)v.y;
      xbf[r][c + 2] = (__bf16)v.z;
      xbf[r][c + 3] = (__bf16)v.w;
    }
    if (t + 1 < T_STEPS) {
      __builtin_prefetch(&x[(size_t)(t + 1) * BATCH * IN_DIM +
                            (size_t)rowbase * IN_DIM + tid * 4], 0, 1);
      __builtin_prefetch(&KCbf[(size_t)(t + 1) * HD * HD + tid * 32], 0, 1);
    }
    __syncthreads();  // B0: ring/x writes of prev step visible

    // ---- recurrent pre-activation: x @ W_ih^T + hidden @ W_hh^T + biases
    v8f acc;
#pragma unroll
    for (int r = 0; r < 8; ++r) acc[r] = biasH;
    {
      v16bf a;
#pragma unroll
      for (int p = 0; p < 8; ++p) {
        v2bf q = *(const v2bf*)&xbf[ln][kpair[p]];
        a[2 * p] = q[0];
        a[2 * p + 1] = q[1];
      }
      acc = __builtin_amdgcn_wmma_f32_16x16x32_bf16(false, a, false, Bih, (short)0, acc,
                                                    false, false);
    }
#pragma unroll
    for (int kt = 0; kt < 2; ++kt) {
      v16bf a;
#pragma unroll
      for (int p = 0; p < 8; ++p) {
        v2bf q = *(const v2bf*)&ringbf[hslot][ln][kt * 32 + kpair[p]];
        a[2 * p] = q[0];
        a[2 * p + 1] = q[1];
      }
      acc = __builtin_amdgcn_wmma_f32_16x16x32_bf16(false, a, false, Bhh[kt], (short)0, acc,
                                                    false, false);
    }

    // ---- relu + leaky CTRNN integration (fp32 master state)
    v8f hobs;
#pragma unroll
    for (int r = 0; r < 8; ++r) {
      float h = acc[r] > 0.f ? acc[r] : 0.f;
      float hid = ring32[hslot][r + half * 8][nb + ln];
      hobs[r] = hid * (1.f - ALPHA) + h * ALPHA;
    }

    v8f hnew = hobs;
    if (t >= NHIST) {  // uniform branch across the workgroup
      // slot of h_obs[t-5+j] = (t+j)%5, via incremental wslot
      int slotj[NHIST];
#pragma unroll
      for (int j = 0; j < NHIST; ++j) {
        int s = wslot + j;
        slotj[j] = (s >= NHIST) ? (s - NHIST) : s;
      }
      // h_mlp = hist(5*H) @ W_mlp^T + b_mlp   (10 K-tiles over ring slots)
      v8f am;
#pragma unroll
      for (int r = 0; r < 8; ++r) am[r] = biasM;
#pragma unroll
      for (int f = 0; f < 10; ++f) {
        v16bf a;
#pragma unroll
        for (int p = 0; p < 8; ++p) {
          int kk = f * 32 + kpair[p];
          int j = kk >> 6;                   // history age (oldest first)
          int hh = kk & 63;
          v2bf q = *(const v2bf*)&ringbf[slotj[j]][ln][hh];
          a[2 * p] = q[0];
          a[2 * p + 1] = q[1];
        }
        am = __builtin_amdgcn_wmma_f32_16x16x32_bf16(false, a, false, Bmlp[f], (short)0, am,
                                                     false, false);
      }
      // D = h_mlp - h_obs  -> LDS (all 64 columns needed as K by every wave)
#pragma unroll
      for (int r = 0; r < 8; ++r)
        dtile[r + half * 8][nb + ln] = (__bf16)(am[r] - hobs[r]);
      __syncthreads();  // B1: D complete; also orders HIST/x reads vs commit
      // h_new = h_obs + D @ KC_t^T  (KC fragments streamed from L2-resident ws)
#pragma unroll
      for (int kt = 0; kt < 2; ++kt) {
        v16bf a, bkc;
#pragma unroll
        for (int p = 0; p < 8; ++p) {
          int k = kt * 32 + kpair[p];
          v2bf q = *(const v2bf*)&dtile[ln][k];
          a[2 * p] = q[0];
          a[2 * p + 1] = q[1];
          v2bf qb = *(const v2bf*)&KCbf[(size_t)t * HD * HD + (size_t)(nb + ln) * HD + k];
          bkc[2 * p] = qb[0];
          bkc[2 * p + 1] = qb[1];
        }
        hnew = __builtin_amdgcn_wmma_f32_16x16x32_bf16(false, a, false, bkc, (short)0, hnew,
                                                       false, false);
      }
    }

    // ---- commit: ring (fp32 + bf16 mirror) and streamed output
#pragma unroll
    for (int r = 0; r < 8; ++r) {
      int m = r + half * 8;
      ring32[wslot][m][nb + ln] = hnew[r];
      ringbf[wslot][m][nb + ln] = (__bf16)hnew[r];
      out[(size_t)t * BATCH * HD + (size_t)(rowbase + m) * HD + nb + ln] = hnew[r];
      if (t == T_STEPS - 1)
        hlast[(size_t)(rowbase + m) * HD + nb + ln] = hnew[r];
    }
    if (t < NHIST) __syncthreads();  // early steps lack the mid-step barrier
    wslot = (wslot == NHIST - 1) ? 0 : wslot + 1;
  }
}

extern "C" void kernel_launch(void* const* d_in, const int* in_sizes, int n_in,
                              void* d_out, int out_size, void* d_ws, size_t ws_size,
                              hipStream_t stream) {
  (void)in_sizes; (void)n_in; (void)out_size; (void)ws_size;
  const float* x     = (const float*)d_in[0];
  const float* W_ih  = (const float*)d_in[1];
  const float* b_ih  = (const float*)d_in[2];
  const float* W_hh  = (const float*)d_in[3];
  const float* b_hh  = (const float*)d_in[4];
  const float* Cm    = (const float*)d_in[5];
  const float* W_mlp = (const float*)d_in[6];
  const float* b_mlp = (const float*)d_in[7];

  float* out   = (float*)d_out;
  float* hlast = out + (size_t)T_STEPS * BATCH * HD;
  unsigned short* KCws = (unsigned short*)d_ws;  // [T][64][64] bf16 (8.2 MB)

  hipLaunchKernelGGL(kc_precompute, dim3(1), dim3(256), 0, stream, W_hh, Cm, KCws);
  hipLaunchKernelGGL(ctrnn_main, dim3(BATCH / 16), dim3(128), 0, stream,
                     x, W_ih, b_ih, W_hh, b_hh, W_mlp, b_mlp, KCws, out, hlast);
}
